// Sonata3DSegLevelEncoder_171798692047
// MI455X (gfx1250) — matrix operand order repaired and checked by reading, not compiled, not run on hardware
//
#include <hip/hip_runtime.h>
#include <hip/hip_bf16.h>

#define NB 2      // batch
#define SEG 512   // max segments
#define HID 768

typedef __attribute__((ext_vector_type(16))) __bf16 v16bf;
typedef __attribute__((ext_vector_type(8)))  float  v8f;
typedef __attribute__((ext_vector_type(4)))  int    v4i;

#if defined(__has_builtin)
#if __has_builtin(__builtin_amdgcn_global_load_async_to_lds_b128)
#define HAVE_ASYNC_LDS 1
#endif
#endif

#ifdef HAVE_ASYNC_LDS
typedef __attribute__((address_space(1))) v4i gv4i;   // global int4
typedef __attribute__((address_space(3))) v4i lv4i;   // LDS int4
__device__ __forceinline__ void async_wait0() {
#if __has_builtin(__builtin_amdgcn_s_wait_asynccnt)
    __builtin_amdgcn_s_wait_asynccnt(0);
#else
    asm volatile("s_wait_asynccnt 0x0" ::: "memory");
#endif
}
#endif

// ---------------------------------------------------------------- index maps
__global__ void k_copy_map(const int* __restrict__ inv, int* __restrict__ out, int total) {
    int i = blockIdx.x * blockDim.x + threadIdx.x;
    if (i < total) out[i] = inv[i];
}

// out[b,n] = parent[b, inv[b,n]]  (compose inverse maps down the cascade)
__global__ void k_compose_map(const int* __restrict__ inv, const int* __restrict__ parent,
                              int* __restrict__ out, int Ni, int Nprev, int total) {
    int i = blockIdx.x * blockDim.x + threadIdx.x;
    if (i >= total) return;
    int b = i / Ni;
    out[i] = parent[b * Nprev + inv[i]];
}

// ---------------------------------------------------------------- scatter-mean
__global__ void k_count(const int* __restrict__ p2s, float* __restrict__ cnt, int Ni, int total) {
    int i = blockIdx.x * blockDim.x + threadIdx.x;
    if (i >= total) return;
    int b = i / Ni;
    unsafeAtomicAdd(&cnt[b * SEG + p2s[i]], 1.0f);
}

// One block per point; scatter-add one source level's native features into segsum columns.
// float4 gather reads (global_load_b128) + scalar f32 atomics into the segment row.
__global__ void k_scatter(const float* __restrict__ xsrc, const int* __restrict__ p2s,
                          const int* __restrict__ gmap, float* __restrict__ segsum,
                          int Ni, int Nk, int Cnat, int coff, int cfp) {
    int p = blockIdx.x;             // flat point over [B, Ni]
    int b = p / Ni;
    int nl = p - b * Ni;
    int s = p2s[p];
    int idx = gmap ? gmap[p] : nl;
    const float4* src = (const float4*)(xsrc + ((size_t)b * Nk + idx) * Cnat);
    float* dst = segsum + ((size_t)(b * SEG + s)) * cfp + coff;
    int nchunk = Cnat >> 2;
    for (int c = threadIdx.x; c < nchunk; c += blockDim.x) {
        float4 v = src[c];
        unsafeAtomicAdd(dst + 4 * c + 0, v.x);
        unsafeAtomicAdd(dst + 4 * c + 1, v.y);
        unsafeAtomicAdd(dst + 4 * c + 2, v.z);
        unsafeAtomicAdd(dst + 4 * c + 3, v.w);
    }
}

__global__ void k_mean_bf16(const float* __restrict__ segsum, const float* __restrict__ cnt,
                            __bf16* __restrict__ A, int cf, int cfp, int total) {
    int i = blockIdx.x * blockDim.x + threadIdx.x;
    if (i >= total) return;
    int r = i / cfp;
    int c = i - r * cfp;
    float v = 0.0f;
    if (c < cf) v = segsum[i] / fmaxf(cnt[r], 1.0f);
    A[i] = (__bf16)v;
}

// W [C, HID] f32 -> Wbt [HID, cfp] bf16 (K contiguous per output column)
__global__ void k_wt_bf16(const float* __restrict__ W, __bf16* __restrict__ Wbt,
                          int cf, int cfp, int total) {
    int i = blockIdx.x * blockDim.x + threadIdx.x;
    if (i >= total) return;
    int h = i / cfp;
    int c = i - h * cfp;
    Wbt[i] = (__bf16)((c < cf) ? W[(size_t)c * HID + h] : 0.0f);
}

// ---------------------------------------------------------------- WMMA GEMM
// proj[M, N] = sum_k A[M,k] * Wbt[N,k] + bias[N]
// block: 128 threads = 4 waves; tile 32(M) x 64(N); K step 64 via LDS staging.
// Each wave: one 16-wide N slice, two 16-row M accumulators -> 4 WMMA per barrier.
#define TM 32
#define TN 64
#define TK 64
__global__ __launch_bounds__(128) void k_gemm_wmma(const __bf16* __restrict__ A,
                                                   const __bf16* __restrict__ Wbt,
                                                   const float* __restrict__ bias,
                                                   float* __restrict__ proj, int cfp) {
    __shared__ __bf16 lA[TM * TK];   // 32x64 bf16 = 4 KB
    __shared__ __bf16 lB[TN * TK];   // 64x64 bf16 = 8 KB
    const int tid  = threadIdx.x;
    const int lane = tid & 31;
    const int wave = tid >> 5;        // 0..3 -> N sub-tile
    const int half = lane >> 4;       // lane half, selects K/M group per ISA layout
    const int l15  = lane & 15;
    const int r0 = blockIdx.x * TM;
    const int h0 = blockIdx.y * TN;

    const uint4* gA = (const uint4*)A;     // 8 bf16 per uint4
    const uint4* gB = (const uint4*)Wbt;
    uint4* sA = (uint4*)lA;
    uint4* sB = (uint4*)lB;
    const int cfp8 = cfp >> 3;             // uint4 per K row

    // per-thread staging coordinates (compile-time trip counts: 2 A + 4 B vectors)
    const int arow0 = tid >> 3,            acq = tid & 7;    // A d = tid, tid+128
    const int arow1 = (tid + 128) >> 3;
    const int brow0 = tid >> 3;                               // B d = tid + 128*j
    const int brow1 = (tid + 128) >> 3;
    const int brow2 = (tid + 256) >> 3;
    const int brow3 = (tid + 384) >> 3;

    v8f acc0 = {}, acc1 = {};
    for (int kk = 0; kk < cfp; kk += TK) {
        __syncthreads();                   // previous iteration's LDS reads done
        int kq = kk >> 3;
        const uint4* pa0 = gA + (size_t)(r0 + arow0) * cfp8 + kq + acq;
        const uint4* pa1 = gA + (size_t)(r0 + arow1) * cfp8 + kq + acq;
        const uint4* pb0 = gB + (size_t)(h0 + brow0) * cfp8 + kq + acq;
        const uint4* pb1 = gB + (size_t)(h0 + brow1) * cfp8 + kq + acq;
        const uint4* pb2 = gB + (size_t)(h0 + brow2) * cfp8 + kq + acq;
        const uint4* pb3 = gB + (size_t)(h0 + brow3) * cfp8 + kq + acq;
#ifdef HAVE_ASYNC_LDS
        // DMA straight into LDS, no VGPR round-trip; tracked on ASYNCcnt.
        __builtin_amdgcn_global_load_async_to_lds_b128((gv4i*)pa0, (lv4i*)(sA + tid),       0, 0);
        __builtin_amdgcn_global_load_async_to_lds_b128((gv4i*)pa1, (lv4i*)(sA + tid + 128), 0, 0);
        __builtin_amdgcn_global_load_async_to_lds_b128((gv4i*)pb0, (lv4i*)(sB + tid),       0, 0);
        __builtin_amdgcn_global_load_async_to_lds_b128((gv4i*)pb1, (lv4i*)(sB + tid + 128), 0, 0);
        __builtin_amdgcn_global_load_async_to_lds_b128((gv4i*)pb2, (lv4i*)(sB + tid + 256), 0, 0);
        __builtin_amdgcn_global_load_async_to_lds_b128((gv4i*)pb3, (lv4i*)(sB + tid + 384), 0, 0);
        async_wait0();
#else
        // issue all global loads, single wait, then LDS stores
        uint4 ra0 = *pa0, ra1 = *pa1;
        uint4 rb0 = *pb0, rb1 = *pb1, rb2 = *pb2, rb3 = *pb3;
        sA[tid]       = ra0;
        sA[tid + 128] = ra1;
        sB[tid]       = rb0;
        sB[tid + 128] = rb1;
        sB[tid + 256] = rb2;
        sB[tid + 384] = rb3;
#endif
        __syncthreads();
#pragma unroll
        for (int ks = 0; ks < 2; ++ks) {
            int kbase = ks * 32;
            v16bf a0, a1, bv;
#pragma unroll
            for (int v = 0; v < 8; ++v) {
                // A 16x32 bf16 layout (ISA 7.12.2): lane row M=l15; K per VGPR:
                int ka = kbase + ((v >> 2) * 16) + half * 8 + (v & 3) * 2;
                a0[2 * v]     = lA[l15 * TK + ka];
                a0[2 * v + 1] = lA[l15 * TK + ka + 1];
                a1[2 * v]     = lA[(16 + l15) * TK + ka];
                a1[2 * v + 1] = lA[(16 + l15) * TK + ka + 1];
                // B 32x16 bf16 layout: col N=l15; K per VGPR same as A
                int kb = kbase + ((v >> 2) * 16) + half * 8 + (v & 3) * 2;
                bv[2 * v]     = lB[(wave * 16 + l15) * TK + kb];
                bv[2 * v + 1] = lB[(wave * 16 + l15) * TK + kb + 1];
            }
            acc0 = __builtin_amdgcn_wmma_f32_16x16x32_bf16(false, a0, false, bv,
                                                           (short)0, acc0, false, false);
            acc1 = __builtin_amdgcn_wmma_f32_16x16x32_bf16(false, a1, false, bv,
                                                           (short)0, acc1, false, false);
        }
    }
    // D layout: VGPR r -> row M = half*8 + r, col N = l15
    int col = h0 + wave * 16 + l15;
    float bsc = bias[col];
#pragma unroll
    for (int r = 0; r < 8; ++r) {
        int row = r0 + half * 8 + r;
        proj[(size_t)row * HID + col]        = acc0[r] + bsc;
        proj[(size_t)(row + 16) * HID + col] = acc1[r] + bsc;
    }
}

// ---------------------------------------------------------------- LayerNorm
__global__ __launch_bounds__(256) void k_ln(const float* __restrict__ proj,
                                            const float* __restrict__ g,
                                            const float* __restrict__ be,
                                            float* __restrict__ out) {
    int r = blockIdx.x;
    const float* x = proj + (size_t)r * HID;
    float s = 0.f, s2 = 0.f;
    for (int c = threadIdx.x; c < HID; c += 256) {
        float v = x[c];
        s += v; s2 += v * v;
    }
    __shared__ float sh[256], sh2[256];
    sh[threadIdx.x] = s; sh2[threadIdx.x] = s2;
    __syncthreads();
    for (int o = 128; o > 0; o >>= 1) {
        if (threadIdx.x < o) {
            sh[threadIdx.x]  += sh[threadIdx.x + o];
            sh2[threadIdx.x] += sh2[threadIdx.x + o];
        }
        __syncthreads();
    }
    float mean = sh[0] * (1.0f / HID);
    float var  = sh2[0] * (1.0f / HID) - mean * mean;
    float rstd = rsqrtf(var + 1e-5f);
    float* o_ = out + (size_t)r * HID;
    for (int c = threadIdx.x; c < HID; c += 256)
        o_[c] = (x[c] - mean) * rstd * g[c] + be[c];
}

// ---------------------------------------------------------------- host
extern "C" void kernel_launch(void* const* d_in, const int* in_sizes, int n_in,
                              void* d_out, int out_size, void* d_ws, size_t ws_size,
                              hipStream_t stream) {
    static const int NPTS[5]  = {256, 1024, 4096, 16384, 65536};
    static const int CNAT[5]  = {512, 384, 192, 96, 48};
    static const int CFULL[5] = {512, 896, 1088, 1184, 1232};
    static const int CFPAD[5] = {512, 896, 1088, 1216, 1280};  // pad K to 64
    const int CFPMAX = 1280;

    const float* x[5]; const int* p2s[5]; const float* W[5];
    const float* bb[5]; const float* g[5]; const float* be[5];
    for (int i = 0; i < 5; ++i) {
        x[i]   = (const float*)d_in[6 * i + 0];
        p2s[i] = (const int*)  d_in[6 * i + 1];
        W[i]   = (const float*)d_in[6 * i + 2];
        bb[i]  = (const float*)d_in[6 * i + 3];
        g[i]   = (const float*)d_in[6 * i + 4];
        be[i]  = (const float*)d_in[6 * i + 5];
    }
    const int* inv[5] = {nullptr, (const int*)d_in[30], (const int*)d_in[31],
                         (const int*)d_in[32], (const int*)d_in[33]};
    float* out = (float*)d_out;
    char* ws = (char*)d_ws;

    // workspace layout
    size_t off = 0;
    int mapOff[5][5];
    for (int i = 1; i < 5; ++i)
        for (int k = i - 1; k >= 0; --k) {
            mapOff[i][k] = (int)(off / 4);
            off += (size_t)NB * NPTS[i] * sizeof(int);
        }
    off = (off + 255) & ~(size_t)255;
    int* maps = (int*)ws;
    float* segsum = (float*)(ws + off); off += (size_t)NB * SEG * CFPMAX * 4;
    float* cnt    = (float*)(ws + off); off += (size_t)NB * SEG * 4;
    off = (off + 255) & ~(size_t)255;
    __bf16* Abuf  = (__bf16*)(ws + off); off += (size_t)NB * SEG * CFPMAX * 2;
    __bf16* Wbt   = (__bf16*)(ws + off); off += (size_t)HID * CFPMAX * 2;
    off = (off + 255) & ~(size_t)255;
    float* proj   = (float*)(ws + off); off += (size_t)NB * SEG * HID * 4;
    (void)ws_size; (void)in_sizes; (void)n_in; (void)out_size;

    // 1) compose inverse maps: map[i][k][b,n] = index into level k for point n of level i
    for (int i = 1; i < 5; ++i) {
        int total = NB * NPTS[i];
        int grid = (total + 255) / 256;
        k_copy_map<<<grid, 256, 0, stream>>>(inv[i], maps + mapOff[i][i - 1], total);
        for (int k = i - 2; k >= 0; --k)
            k_compose_map<<<grid, 256, 0, stream>>>(inv[i], maps + mapOff[i - 1][k],
                                                    maps + mapOff[i][k],
                                                    NPTS[i], NPTS[i - 1], total);
    }

    // 2) per-level: scatter-mean -> bf16 GEMM (WMMA) -> LayerNorm
    for (int i = 0; i < 5; ++i) {
        int cfp = CFPAD[i], cf = CFULL[i];
        (void)hipMemsetAsync(segsum, 0, (size_t)NB * SEG * cfp * 4, stream);
        (void)hipMemsetAsync(cnt,    0, (size_t)NB * SEG * 4, stream);

        int totalPts = NB * NPTS[i];
        k_count<<<(totalPts + 255) / 256, 256, 0, stream>>>(p2s[i], cnt, NPTS[i], totalPts);

        int coff = 0;
        for (int k = i; k >= 0; --k) {   // concat order: [x_i | x_{i-1} | ... | x_0]
            const int* gmap = (k == i) ? nullptr : (maps + mapOff[i][k]);
            k_scatter<<<totalPts, 128, 0, stream>>>(x[k], p2s[i], gmap, segsum,
                                                    NPTS[i], NPTS[k], CNAT[k], coff, cfp);
            coff += CNAT[k];
        }

        int totalA = NB * SEG * cfp;
        k_mean_bf16<<<(totalA + 255) / 256, 256, 0, stream>>>(segsum, cnt, Abuf, cf, cfp, totalA);
        int totalW = HID * cfp;
        k_wt_bf16<<<(totalW + 255) / 256, 256, 0, stream>>>(W[i], Wbt, cf, cfp, totalW);

        dim3 gg(NB * SEG / TM, HID / TN);
        k_gemm_wmma<<<gg, 128, 0, stream>>>(Abuf, Wbt, bb[i], proj, cfp);

        k_ln<<<NB * SEG, 256, 0, stream>>>(proj, g[i], be[i],
                                           out + (size_t)i * NB * SEG * HID);
    }
}
